// MoEVisionTransformer_37933151158755
// MI455X (gfx1250) — matrix operand order repaired and checked by reading, not compiled, not run on hardware
//
#include <hip/hip_runtime.h>
#include <hip/hip_bf16.h>

// ---------------------------------------------------------------------------
// MoE ViT forward for gfx1250 (MI455X). All GEMMs use v_wmma_f32_16x16x32_bf16.
// Branch-free WMMA inner loop; each wave computes a 16x64 strip (4 WMMA
// accumulators share one A fragment per K-step; all 10 fragment loads issued
// before the first WMMA so waits are partial, not full drains).
// Workspace requirement: ~860 MB (bf16 weight mirrors + activations).
// ---------------------------------------------------------------------------

typedef __attribute__((ext_vector_type(16))) __bf16 v16bf;
typedef __attribute__((ext_vector_type(8)))  float  v8f;

#define DDIM   768
#define NHEAD  12
#define NDEPTH 12
#define NEXP   4
#define HIDD   3072
#define NTOK   197
#define NPAD   208   // padded token dim (13*16) for attention M/N
#define KPAD   224   // padded token dim (7*32) for attention K (P@V)
#define BS     32
#define NPATCH 196
#define MTOT   (BS * NTOK)    // 6304 = 394 * 16
#define MPATCH (BS * NPATCH)  // 6272
#define HD     64
#define NCLASS 1000

union FragU { uint4 u[2]; v16bf v; };

// ---------------------------------------------------------------------------
// Generic WMMA GEMM:  C[z] = act(A[z] (M,K) * W[z] (N,K)^T + bias) + R
// 256 threads = 8 waves stacked along M; wave tile 16(M) x 64(N) via 4 accs;
// block tile 128 x 64. REQUIRES K % 32 == 0. Out-of-range rows/cols are
// clamped on load (their lanes discarded at store): branch-free K loop.
// ---------------------------------------------------------------------------
__global__ __launch_bounds__(256)
void wmma_gemm(const __hip_bfloat16* __restrict__ A, long aStride, int lda,
               const __hip_bfloat16* __restrict__ W, long wStride, int ldw,
               const float* __restrict__ bias, long bStride,
               const float* __restrict__ R, long rStride, int ldr,
               float* __restrict__ Cf, __hip_bfloat16* __restrict__ Cb,
               int zdiv, long cS1, long cS2, int ldc,
               const int* __restrict__ rowMap, long rmStride,
               const int* __restrict__ mCount,
               int M, int N, int K, int act)
{
    const int z    = blockIdx.z;
    const int lane = threadIdx.x & 31;
    const int wave = threadIdx.x >> 5;

    int Mlim = M;
    if (mCount) { int c = mCount[z]; Mlim = (c < M) ? c : M; }

    const int tm = blockIdx.x * 128 + wave * 16;         // wave row base
    const int tn = blockIdx.y * 64;                      // wave col base
    if (tm >= Mlim || tn >= N) return;                   // wave-uniform exit

    const __hip_bfloat16* Az = A + (size_t)z * aStride;
    const __hip_bfloat16* Wz = W + (size_t)z * wStride;

    const int g   = lane >> 4;      // lane group 0/1
    const int l16 = lane & 15;

    int am = tm + l16; if (am >= Mlim) am = Mlim - 1;    // clamp: result discarded

    // A fragment base pointers (16B aligned: lda/ldw multiples of 8).
    const uint4* pa0 = (const uint4*)(Az + (size_t)am * lda + g * 8);
    const uint4* pa1 = (const uint4*)(Az + (size_t)am * lda + g * 8 + 16);
    // 4 B-fragment base pointers (cols tn+s*16+l16, clamped)
    int bn0 = tn +  0 + l16; if (bn0 >= N) bn0 = N - 1;
    int bn1 = tn + 16 + l16; if (bn1 >= N) bn1 = N - 1;
    int bn2 = tn + 32 + l16; if (bn2 >= N) bn2 = N - 1;
    int bn3 = tn + 48 + l16; if (bn3 >= N) bn3 = N - 1;
    const uint4* pw0 = (const uint4*)(Wz + (size_t)bn0 * ldw + g * 8);
    const uint4* pw1 = (const uint4*)(Wz + (size_t)bn1 * ldw + g * 8);
    const uint4* pw2 = (const uint4*)(Wz + (size_t)bn2 * ldw + g * 8);
    const uint4* pw3 = (const uint4*)(Wz + (size_t)bn3 * ldw + g * 8);

    v8f acc0 = {0.f, 0.f, 0.f, 0.f, 0.f, 0.f, 0.f, 0.f};
    v8f acc1 = acc0, acc2 = acc0, acc3 = acc0;

    const int steps = K >> 5;
    for (int it = 0; it < steps; ++it) {
        // Issue ALL fragment loads first (distinct live values -> distinct
        // VGPRs -> partial s_wait_loadcnt before each WMMA, not full drains).
        FragU af, b0, b1, b2, b3;
        af.u[0] = pa0[it * 4];                           // K {k0+g*8 .. +7}
        af.u[1] = pa1[it * 4];                           // K {k0+16+g*8 .. +7}
        b0.u[0] = pw0[it * 4]; b0.u[1] = pw0[it * 4 + 2];// K {k0+g*8 .. +7}, {k0+16+g*8 .. +7}
        b1.u[0] = pw1[it * 4]; b1.u[1] = pw1[it * 4 + 2];
        b2.u[0] = pw2[it * 4]; b2.u[1] = pw2[it * 4 + 2];
        b3.u[0] = pw3[it * 4]; b3.u[1] = pw3[it * 4 + 2];
        acc0 = __builtin_amdgcn_wmma_f32_16x16x32_bf16(
                   false, af.v, false, b0.v, (short)0, acc0, false, false);
        acc1 = __builtin_amdgcn_wmma_f32_16x16x32_bf16(
                   false, af.v, false, b1.v, (short)0, acc1, false, false);
        acc2 = __builtin_amdgcn_wmma_f32_16x16x32_bf16(
                   false, af.v, false, b2.v, (short)0, acc2, false, false);
        acc3 = __builtin_amdgcn_wmma_f32_16x16x32_bf16(
                   false, af.v, false, b3.v, (short)0, acc3, false, false);
    }

    // Store: acc[s] VGPR r -> row tm + g*8 + r, col tn + s*16 + l16.
    const long cOff = (long)(z / zdiv) * cS1 + (long)(z % zdiv) * cS2;
    const int* rm   = rowMap ? (rowMap + (size_t)z * rmStride) : nullptr;

    v8f accs[4] = {acc0, acc1, acc2, acc3};
#pragma unroll
    for (int s = 0; s < 4; ++s) {
        const int col = tn + s * 16 + l16;
        if (col >= N) continue;                          // per-lane; WMMA done
        const float bval = bias ? bias[(size_t)z * bStride + col] : 0.0f;
#pragma unroll
        for (int r = 0; r < 8; ++r) {
            const int row = tm + g * 8 + r;
            if (row >= Mlim) continue;
            const int orow = rm ? rm[row] : row;
            float v = accs[s][r] + bval;
            if (act == 1) v = 0.5f * v * (1.0f + erff(v * 0.70710678118f)); // GELU
            if (R) v += R[(size_t)z * rStride + (size_t)orow * ldr + col];
            if (Cf) Cf[cOff + (size_t)orow * ldc + col] = v;
            else    Cb[cOff + (size_t)orow * ldc + col] = __float2bfloat16(v);
        }
    }
}

// ---------------------------------------------------------------------------
// Support kernels
// ---------------------------------------------------------------------------
__global__ void cvt_f32_bf16_k(const float* __restrict__ src,
                               __hip_bfloat16* __restrict__ dst, long n)
{
    long i = (long)blockIdx.x * blockDim.x + threadIdx.x;
    long stride = (long)gridDim.x * blockDim.x;
    for (; i < n; i += stride) dst[i] = __float2bfloat16(src[i]);
}

// im2col: x[32,3,224,224] -> [6272, 768] bf16, inner order (c, ph, pw)
__global__ void im2col_k(const float* __restrict__ x,
                         __hip_bfloat16* __restrict__ out)
{
    long i = (long)blockIdx.x * blockDim.x + threadIdx.x;
    const long total = (long)MPATCH * DDIM;
    if (i >= total) return;
    const int  col = (int)(i % DDIM);
    const long row = i / DDIM;
    const int b = (int)(row / NPATCH);
    const int p = (int)(row % NPATCH);
    const int py = p / 14, px = p % 14;
    const int c  = col >> 8;
    const int ph = (col >> 4) & 15;
    const int pw = col & 15;
    const float v = x[(((long)b * 3 + c) * 224 + (py * 16 + ph)) * 224 + (px * 16 + pw)];
    out[i] = __float2bfloat16(v);
}

// x[b,0,:] = cls + pos[0];  x[b,1+p,:] = patch[b*196+p] + pos[1+p]
__global__ void assemble_x_k(const float* __restrict__ patch,
                             const float* __restrict__ cls_tok,
                             const float* __restrict__ pos,
                             float* __restrict__ xo)
{
    long i = (long)blockIdx.x * blockDim.x + threadIdx.x;
    const long total = (long)MTOT * DDIM;
    if (i >= total) return;
    const int d = (int)(i % DDIM);
    const int n = (int)((i / DDIM) % NTOK);
    const int b = (int)(i / ((long)DDIM * NTOK));
    float v = (n == 0) ? cls_tok[d]
                       : patch[((long)b * NPATCH + (n - 1)) * DDIM + d];
    xo[i] = v + pos[(size_t)n * DDIM + d];
}

__global__ __launch_bounds__(256)
void layernorm_k(const float* __restrict__ x, const float* __restrict__ w,
                 const float* __restrict__ b, __hip_bfloat16* __restrict__ y)
{
    const int row = blockIdx.x;
    const float* xr = x + (size_t)row * DDIM;
    __shared__ float s1[256], s2[256];
    float a = 0.f, q = 0.f;
    for (int j = threadIdx.x; j < DDIM; j += 256) { float v = xr[j]; a += v; q += v * v; }
    s1[threadIdx.x] = a; s2[threadIdx.x] = q;
    __syncthreads();
    for (int off = 128; off > 0; off >>= 1) {
        if (threadIdx.x < off) {
            s1[threadIdx.x] += s1[threadIdx.x + off];
            s2[threadIdx.x] += s2[threadIdx.x + off];
        }
        __syncthreads();
    }
    const float mean = s1[0] * (1.0f / DDIM);
    const float var  = s2[0] * (1.0f / DDIM) - mean * mean;
    const float inv  = rsqrtf(var + 1e-5f);
    for (int j = threadIdx.x; j < DDIM; j += 256)
        y[(size_t)row * DDIM + j] = __float2bfloat16((xr[j] - mean) * inv * w[j] + b[j]);
}

// qkv[6304,2304] -> q/k padded [B*H,208,64] bf16 (q *= 1/8), v^T [B*H,64,224]
__global__ void qkv_split_k(const float* __restrict__ qkv,
                            __hip_bfloat16* __restrict__ qb,
                            __hip_bfloat16* __restrict__ kb,
                            __hip_bfloat16* __restrict__ vt)
{
    long i = (long)blockIdx.x * blockDim.x + threadIdx.x;
    const long total = (long)BS * NHEAD * KPAD * HD;   // iterate widest (224)
    if (i >= total) return;
    const int  d  = (int)(i & 63);
    const long r  = i >> 6;
    const int  n  = (int)(r % KPAD);
    const long bh = r / KPAD;           // b*12 + h
    const int  h  = (int)(bh % NHEAD);
    const int  b  = (int)(bh / NHEAD);
    float q = 0.f, k = 0.f, v = 0.f;
    if (n < NTOK) {
        const size_t t = ((size_t)b * NTOK + n) * (3 * DDIM) + (size_t)h * HD + d;
        q = qkv[t] * 0.125f;            // 1/sqrt(64)
        k = qkv[t + DDIM];
        v = qkv[t + 2 * DDIM];
    }
    if (n < NPAD) {
        const size_t o = ((size_t)bh * NPAD + n) * HD + d;
        qb[o] = __float2bfloat16(q);
        kb[o] = __float2bfloat16(k);
    }
    vt[((size_t)bh * HD + d) * KPAD + n] = __float2bfloat16(v);
}

// softmax over 197 valid cols of scores[z,q,0..207]; bf16 probs [z,208,224]
__global__ __launch_bounds__(256)
void softmax_k(const float* __restrict__ sc, __hip_bfloat16* __restrict__ pr)
{
    const int q = blockIdx.x;
    const int z = blockIdx.y;
    const float* row = sc + ((size_t)z * NPAD + q) * NPAD;
    __hip_bfloat16* prow = pr + ((size_t)z * NPAD + q) * KPAD;
    __shared__ float sm[256];
    float mx = -1e30f;
    for (int j = threadIdx.x; j < NTOK; j += 256) mx = fmaxf(mx, row[j]);
    sm[threadIdx.x] = mx; __syncthreads();
    for (int off = 128; off > 0; off >>= 1) {
        if (threadIdx.x < off) sm[threadIdx.x] = fmaxf(sm[threadIdx.x], sm[threadIdx.x + off]);
        __syncthreads();
    }
    mx = sm[0]; __syncthreads();
    float s = 0.f;
    for (int j = threadIdx.x; j < NTOK; j += 256) s += expf(row[j] - mx);
    sm[threadIdx.x] = s; __syncthreads();
    for (int off = 128; off > 0; off >>= 1) {
        if (threadIdx.x < off) sm[threadIdx.x] += sm[threadIdx.x + off];
        __syncthreads();
    }
    const float inv = 1.0f / sm[0];
    for (int j = threadIdx.x; j < KPAD; j += 256)
        prow[j] = __float2bfloat16(j < NTOK ? expf(row[j] - mx) * inv : 0.0f);
}

__global__ void zero_ints_k(int* p, int n)
{
    int i = blockIdx.x * blockDim.x + threadIdx.x;
    if (i < n) p[i] = 0;
}

// gate logits + argmax routing (softmax over top-1 == 1.0, so hard routing)
__global__ void gate_route_k(const __hip_bfloat16* __restrict__ y,
                             const __hip_bfloat16* __restrict__ gw,
                             const float* __restrict__ gb,
                             int* __restrict__ cnt, int* __restrict__ idx)
{
    const int t = blockIdx.x * blockDim.x + threadIdx.x;
    if (t >= MTOT) return;
    const __hip_bfloat16* yr = y + (size_t)t * DDIM;
    float best = -1e30f; int be = 0;
    for (int e = 0; e < NEXP; ++e) {
        const __hip_bfloat16* wr = gw + (size_t)e * DDIM;
        float s = gb[e];
        for (int j = 0; j < DDIM; ++j)
            s += __bfloat162float(yr[j]) * __bfloat162float(wr[j]);
        if (s > best) { best = s; be = e; }
    }
    const int pos = atomicAdd(&cnt[be], 1);
    idx[be * MTOT + pos] = t;
}

__global__ __launch_bounds__(256)
void gather_rows_k(const __hip_bfloat16* __restrict__ y,
                   const int* __restrict__ cnt, const int* __restrict__ idx,
                   __hip_bfloat16* __restrict__ yg)
{
    const int e = blockIdx.y;
    const int i = blockIdx.x;
    if (i >= cnt[e]) return;
    const int src = idx[e * MTOT + i];
    const __hip_bfloat16* s = y + (size_t)src * DDIM;
    __hip_bfloat16* d = yg + ((size_t)e * MTOT + i) * DDIM;
    for (int j = threadIdx.x; j < DDIM; j += 256) d[j] = s[j];
}

__global__ void gather_cls_k(const __hip_bfloat16* __restrict__ y,
                             __hip_bfloat16* __restrict__ cls)
{
    const int b = blockIdx.x;
    const __hip_bfloat16* s = y + (size_t)b * NTOK * DDIM;
    for (int j = threadIdx.x; j < DDIM; j += blockDim.x)
        cls[(size_t)b * DDIM + j] = s[j];
}

// ---------------------------------------------------------------------------
// Host launcher
// ---------------------------------------------------------------------------
extern "C" void kernel_launch(void* const* d_in, const int* in_sizes, int n_in,
                              void* d_out, int out_size, void* d_ws, size_t ws_size,
                              hipStream_t stream)
{
    (void)in_sizes; (void)n_in; (void)out_size; (void)ws_size;

    const float* in_x      = (const float*)d_in[0];
    const float* conv_w    = (const float*)d_in[1];
    const float* conv_b    = (const float*)d_in[2];
    const float* cls_tok   = (const float*)d_in[3];
    const float* pos_emb   = (const float*)d_in[4];
    const float* ln1_w     = (const float*)d_in[5];
    const float* ln1_b     = (const float*)d_in[6];
    const float* qkv_w     = (const float*)d_in[7];
    const float* qkv_b     = (const float*)d_in[8];
    const float* out_w     = (const float*)d_in[9];
    const float* out_b     = (const float*)d_in[10];
    const float* ln2_w     = (const float*)d_in[11];
    const float* ln2_b     = (const float*)d_in[12];
    const float* gate_w    = (const float*)d_in[13];
    const float* gate_b    = (const float*)d_in[14];
    const float* e1_w      = (const float*)d_in[15];
    const float* e1_b      = (const float*)d_in[16];
    const float* e2_w      = (const float*)d_in[17];
    const float* e2_b      = (const float*)d_in[18];
    const float* fnorm_w   = (const float*)d_in[19];
    const float* fnorm_b   = (const float*)d_in[20];
    const float* head_w    = (const float*)d_in[21];
    const float* head_b    = (const float*)d_in[22];
    float* out             = (float*)d_out;

    // ---- workspace carve-up (256B aligned) ----
    char* ws = (char*)d_ws;
    size_t off = 0;
    auto alloc = [&](size_t bytes) -> char* {
        char* p = ws + off;
        off = (off + bytes + 255) & ~(size_t)255;
        return p;
    };
    typedef __hip_bfloat16 bf16;
    const long SZ_QKVW = (long)NDEPTH * 3 * DDIM * DDIM;
    const long SZ_OUTW = (long)NDEPTH * DDIM * DDIM;
    const long SZ_EW   = (long)NDEPTH * NEXP * HIDD * DDIM;

    bf16* wb_conv = (bf16*)alloc((size_t)DDIM * DDIM * 2);
    bf16* wb_qkv  = (bf16*)alloc((size_t)SZ_QKVW * 2);
    bf16* wb_out  = (bf16*)alloc((size_t)SZ_OUTW * 2);
    bf16* wb_gate = (bf16*)alloc((size_t)NDEPTH * NEXP * DDIM * 2);
    bf16* wb_e1   = (bf16*)alloc((size_t)SZ_EW * 2);
    bf16* wb_e2   = (bf16*)alloc((size_t)SZ_EW * 2);
    bf16* wb_head = (bf16*)alloc((size_t)NCLASS * DDIM * 2);

    bf16*  im2col = (bf16*) alloc((size_t)MPATCH * DDIM * 2);
    float* patch  = (float*)alloc((size_t)MPATCH * DDIM * 4);
    float* xact   = (float*)alloc((size_t)MTOT * DDIM * 4);
    bf16*  ybf    = (bf16*) alloc((size_t)MTOT * DDIM * 2);
    float* qkvf   = (float*)alloc((size_t)MTOT * 3 * DDIM * 4);
    bf16*  qb     = (bf16*) alloc((size_t)BS * NHEAD * NPAD * HD * 2);
    bf16*  kb     = (bf16*) alloc((size_t)BS * NHEAD * NPAD * HD * 2);
    bf16*  vt     = (bf16*) alloc((size_t)BS * NHEAD * HD * KPAD * 2);
    float* scores = (float*)alloc((size_t)BS * NHEAD * NPAD * NPAD * 4);
    bf16*  probs  = (bf16*) alloc((size_t)BS * NHEAD * NPAD * KPAD * 2);
    bf16*  ctxbf  = (bf16*) alloc((size_t)MTOT * DDIM * 2);
    int*   cnt    = (int*)  alloc(NEXP * 4);
    int*   idx    = (int*)  alloc((size_t)NEXP * MTOT * 4);
    bf16*  yg     = (bf16*) alloc((size_t)NEXP * MTOT * DDIM * 2);
    bf16*  h1bf   = (bf16*) alloc((size_t)MTOT * HIDD * 2);
    bf16*  clsbf  = (bf16*) alloc((size_t)BS * DDIM * 2);

    auto cvt = [&](const float* s, bf16* d, long n) {
        long blocks = (n + 255) / 256;
        if (blocks > 65535) blocks = 65535;
        cvt_f32_bf16_k<<<dim3((unsigned)blocks), dim3(256), 0, stream>>>(s, d, n);
    };
    // GEMM launcher: C[z] = act(A[z](M,K) * W[z](N,K)^T + bias) + R
    auto gemm = [&](const bf16* A, long aS, int lda,
                    const bf16* W, long wS, int ldw,
                    const float* bias, long bS,
                    const float* R, long rS, int ldr,
                    float* Cf, bf16* Cb,
                    int zdiv, long cS1, long cS2, int ldc,
                    const int* rowMap, long rmS, const int* mCount,
                    int M, int N, int K, int act, int zcount) {
        dim3 grid((M + 127) / 128, (N + 63) / 64, zcount);
        wmma_gemm<<<grid, dim3(256), 0, stream>>>(A, aS, lda, W, wS, ldw, bias, bS,
                                                  R, rS, ldr, Cf, Cb,
                                                  zdiv, cS1, cS2, ldc,
                                                  rowMap, rmS, mCount, M, N, K, act);
    };

    // ---- weight conversion (bf16 mirrors) ----
    cvt(conv_w, wb_conv, (long)DDIM * DDIM);
    cvt(qkv_w,  wb_qkv,  SZ_QKVW);
    cvt(out_w,  wb_out,  SZ_OUTW);
    cvt(gate_w, wb_gate, (long)NDEPTH * NEXP * DDIM);
    cvt(e1_w,   wb_e1,   SZ_EW);
    cvt(e2_w,   wb_e2,   SZ_EW);
    cvt(head_w, wb_head, (long)NCLASS * DDIM);

    // ---- patch embed ----
    {
        long total = (long)MPATCH * DDIM;
        im2col_k<<<dim3((unsigned)((total + 255) / 256)), dim3(256), 0, stream>>>(in_x, im2col);
        gemm(im2col, 0, DDIM, wb_conv, 0, DDIM, conv_b, 0,
             nullptr, 0, 0, patch, nullptr, 1, 0, 0, DDIM,
             nullptr, 0, nullptr, MPATCH, DDIM, DDIM, 0, 1);
        long t2 = (long)MTOT * DDIM;
        assemble_x_k<<<dim3((unsigned)((t2 + 255) / 256)), dim3(256), 0, stream>>>(
            patch, cls_tok, pos_emb, xact);
    }

    const int ZBH = BS * NHEAD; // 384
    for (int l = 0; l < NDEPTH; ++l) {
        // --- attention ---
        layernorm_k<<<dim3(MTOT), dim3(256), 0, stream>>>(
            xact, ln1_w + (size_t)l * DDIM, ln1_b + (size_t)l * DDIM, ybf);
        gemm(ybf, 0, DDIM, wb_qkv + (size_t)l * 3 * DDIM * DDIM, 0, DDIM,
             qkv_b + (size_t)l * 3 * DDIM, 0, nullptr, 0, 0,
             qkvf, nullptr, 1, 0, 0, 3 * DDIM,
             nullptr, 0, nullptr, MTOT, 3 * DDIM, DDIM, 0, 1);
        {
            long total = (long)ZBH * KPAD * HD;
            qkv_split_k<<<dim3((unsigned)((total + 255) / 256)), dim3(256), 0, stream>>>(
                qkvf, qb, kb, vt);
        }
        // scores[z] = q[z] @ k[z]^T   (batched over 384 (b,h)), K=64
        gemm(qb, (long)NPAD * HD, HD, kb, (long)NPAD * HD, HD,
             nullptr, 0, nullptr, 0, 0, scores, nullptr,
             1, (long)NPAD * NPAD, 0, NPAD,
             nullptr, 0, nullptr, NPAD, NPAD, HD, 0, ZBH);
        softmax_k<<<dim3(NPAD, ZBH), dim3(256), 0, stream>>>(scores, probs);
        // ctx[b, q, h*64+d] = probs[z] @ v[z]  (v transposed, K padded to 224)
        gemm(probs, (long)NPAD * KPAD, KPAD, vt, (long)HD * KPAD, KPAD,
             nullptr, 0, nullptr, 0, 0, nullptr, ctxbf,
             NHEAD, (long)NTOK * DDIM, HD, DDIM,
             nullptr, 0, nullptr, NTOK, HD, KPAD, 0, ZBH);
        // x = x + ctx @ out_w^T + out_b
        gemm(ctxbf, 0, DDIM, wb_out + (size_t)l * DDIM * DDIM, 0, DDIM,
             out_b + (size_t)l * DDIM, 0, xact, 0, DDIM,
             xact, nullptr, 1, 0, 0, DDIM,
             nullptr, 0, nullptr, MTOT, DDIM, DDIM, 0, 1);

        // --- MoE FFN (top-1 => hard routing, weight 1.0) ---
        layernorm_k<<<dim3(MTOT), dim3(256), 0, stream>>>(
            xact, ln2_w + (size_t)l * DDIM, ln2_b + (size_t)l * DDIM, ybf);
        zero_ints_k<<<dim3(1), dim3(32), 0, stream>>>(cnt, NEXP);
        gate_route_k<<<dim3((MTOT + 127) / 128), dim3(128), 0, stream>>>(
            ybf, wb_gate + (size_t)l * NEXP * DDIM, gate_b + (size_t)l * NEXP, cnt, idx);
        gather_rows_k<<<dim3(MTOT, NEXP), dim3(256), 0, stream>>>(ybf, cnt, idx, yg);
        for (int e = 0; e < NEXP; ++e) {
            const long we = ((long)l * NEXP + e) * HIDD * DDIM;
            // h1 = gelu(y_e @ e1^T + b1)   (bf16 out)
            gemm(yg + (size_t)e * MTOT * DDIM, 0, DDIM,
                 wb_e1 + we, 0, DDIM,
                 e1_b + ((size_t)l * NEXP + e) * HIDD, 0,
                 nullptr, 0, 0, nullptr, h1bf,
                 1, 0, 0, HIDD, nullptr, 0, cnt + e,
                 MTOT, HIDD, DDIM, 1, 1);
            // x[token] += h1 @ e2^T + b2   (scatter via row map, fused residual)
            gemm(h1bf, 0, HIDD,
                 wb_e2 + we, 0, HIDD,
                 e2_b + ((size_t)l * NEXP + e) * DDIM, 0,
                 xact, 0, DDIM, xact, nullptr,
                 1, 0, 0, DDIM, idx + e * MTOT, 0, cnt + e,
                 MTOT, DDIM, HIDD, 0, 1);
        }
    }

    // ---- final norm + head ----
    layernorm_k<<<dim3(MTOT), dim3(256), 0, stream>>>(xact, fnorm_w, fnorm_b, ybf);
    gather_cls_k<<<dim3(BS), dim3(256), 0, stream>>>(ybf, clsbf);
    gemm(clsbf, 0, DDIM, wb_head, 0, DDIM, head_b, 0,
         nullptr, 0, 0, out, nullptr, 1, 0, 0, NCLASS,
         nullptr, 0, nullptr, BS, NCLASS, DDIM, 0, 1);
}
